// DotAttentionLayer_36146444763807
// MI455X (gfx1250) — matrix operand order repaired and checked by reading, not compiled, not run on hardware
//
#include <hip/hip_runtime.h>

// ---------------------------------------------------------------------------
// CDNA5 (gfx1250) graph-attention layer, bf16 WMMA path.
//   D = 128, H = 8, HD = 16.  All GEMMs via v_wmma_f32_16x16x32_bf16.
// ---------------------------------------------------------------------------

typedef __attribute__((ext_vector_type(16))) __bf16 v16bf;
typedef __attribute__((ext_vector_type(8)))  float  v8f;

union Frag {
    uint4  q[2];
    v16bf  v;
};

__device__ __forceinline__ unsigned short f2bf_u16(float f) {
    union { float f; unsigned int u; } x; x.f = f;
    unsigned int u = x.u;
    if ((u & 0x7fffffffu) > 0x7f800000u)          // NaN: quiet it
        return (unsigned short)((u >> 16) | 0x0040u);
    return (unsigned short)((u + 0x7fffu + ((u >> 16) & 1u)) >> 16);  // RNE
}

// --------------------------- weight fp32 -> bf16 ---------------------------
__global__ void convert_bf16_kernel(const float* __restrict__ src,
                                    unsigned short* __restrict__ dst, int n) {
    int i = blockIdx.x * blockDim.x + threadIdx.x;
    if (i < n) dst[i] = f2bf_u16(src[i]);
}

__global__ void zero_kernel(float* __restrict__ p, int n) {
    int i = blockIdx.x * blockDim.x + threadIdx.x;
    if (i < n) p[i] = 0.0f;
}

// ---------------------------------------------------------------------------
// Y[n][d] = sum_k X[n][k] * W[d][k] + bias[d]   (optionally ReLU)
// 256 threads = 8 waves. Wave w owns output-column tile w (16 cols).
// Block processes 4 row tiles of 16 rows each (B fragments preloaded once
// into 32 VGPRs and reused across all 4 row tiles).
// ---------------------------------------------------------------------------
__global__ __launch_bounds__(256) void gemm_bias_kernel(
    const float* __restrict__ X, const unsigned short* __restrict__ Wbf,
    const float* __restrict__ bias, float* __restrict__ Y,
    int nrows, int do_relu)
{
    __shared__ __align__(16) unsigned short sA[16 * 136];  // 16x128 bf16, padded

    const int tid  = threadIdx.x;
    const int lane = tid & 31;
    const int ct   = tid >> 5;        // wave id == column tile (0..7)
    const int half = lane >> 4;       // 0|1 -> which K half this lane holds
    const int l16  = lane & 15;

    // ---- Preload B fragments (W rows = output columns), 4 k-tiles of 32 ----
    // B frag (contiguous K order used here, same as A): lane = col, lanes 0-15: K=0..15, 16-31: K=16..31.
    Frag bfr[4];
    {
        const unsigned short* wp = Wbf + (ct * 16 + l16) * 128 + half * 16;
#pragma unroll
        for (int kt = 0; kt < 4; ++kt) {
            const uint4* p = (const uint4*)(wp + kt * 32);
            bfr[kt].q[0] = p[0];
            bfr[kt].q[1] = p[1];
        }
    }
    const float bia = bias[ct * 16 + l16];

    // staging coordinates (each thread moves 8 consecutive elements of a row)
    const int srow = tid >> 4;            // 0..15 row within tile
    const int scol = (tid & 15) * 8;      // 0..120 step 8

    const int rt0 = blockIdx.x * 4;
    for (int rt = 0; rt < 4; ++rt) {
        const int row0 = (rt0 + rt) * 16;
        if (row0 >= nrows) break;                 // uniform within block
        const bool full = (row0 + 16) <= nrows;   // uniform within block

        // ---- stage 16x128 fp32 tile -> bf16 LDS (coalesced float4 loads) ----
        {
            const int grow = row0 + srow;
            float vals[8];
            if (full || grow < nrows) {
                const float4* p = (const float4*)(X + (size_t)grow * 128 + scol);
                float4 a = p[0], b = p[1];
                vals[0]=a.x; vals[1]=a.y; vals[2]=a.z; vals[3]=a.w;
                vals[4]=b.x; vals[5]=b.y; vals[6]=b.z; vals[7]=b.w;
            } else {
#pragma unroll
                for (int j = 0; j < 8; ++j) vals[j] = 0.0f;
            }
            unsigned short* s = sA + srow * 136 + scol;
#pragma unroll
            for (int j = 0; j < 8; ++j) s[j] = f2bf_u16(vals[j]);
        }

        // prefetch next row tile's X data while this tile computes
        {
            const int prow = row0 + 16 + srow;
            if (rt < 3 && prow < nrows)
                __builtin_prefetch(X + (size_t)prow * 128 + scol, 0, 3);
        }
        __syncthreads();

        // ---- 16x16x128 = 4 chained v_wmma_f32_16x16x32_bf16 ----
        v8f acc = {0.f,0.f,0.f,0.f,0.f,0.f,0.f,0.f};
#pragma unroll
        for (int kt = 0; kt < 4; ++kt) {
            // A frag (contiguous K order, matches B): lane = row; lanes 0-15 hold K 0..15,
            // lanes 16-31 hold K 16..31.
            Frag a;
            const unsigned short* ap = sA + l16 * 136 + kt * 32 + half * 16;
            a.q[0] = *(const uint4*)(ap);         // K = kt*32 + half*16 + 0..7
            a.q[1] = *(const uint4*)(ap + 8);     // K = kt*32 + half*16 + 8..15
            acc = __builtin_amdgcn_wmma_f32_16x16x32_bf16(
                      false, a.v, false, bfr[kt].v, (short)0, acc, false, false);
        }

        // ---- epilogue: C VGPR r -> row (r + 8*half), col = ct*16 + l16 ----
        const int colg = ct * 16 + l16;
        float* yp = Y + (size_t)(row0 + half * 8) * 128 + colg;
        if (full) {                                // uniform: straight-line stores
#pragma unroll
            for (int r = 0; r < 8; ++r) {
                float vv = acc[r] + bia;
                if (do_relu) vv = vv > 0.0f ? vv : 0.0f;
                yp[(size_t)r * 128] = vv;
            }
        } else {
#pragma unroll
            for (int r = 0; r < 8; ++r) {
                const int grow = row0 + r + half * 8;
                if (grow < nrows) {
                    float vv = acc[r] + bia;
                    if (do_relu) vv = vv > 0.0f ? vv : 0.0f;
                    Y[(size_t)grow * 128 + colg] = vv;
                }
            }
        }
        __syncthreads();
    }
}

// --------------------- per-(edge,head) scores + denom ----------------------
__global__ void edge_scores_kernel(const float* __restrict__ qp,
                                   const float* __restrict__ kp,
                                   const int* __restrict__ self_idx,
                                   const int* __restrict__ nbr_idx,
                                   float* __restrict__ ex,
                                   float* __restrict__ denom, int E)
{
    int t = blockIdx.x * blockDim.x + threadIdx.x;
    if (t >= E * 8) return;
    const int e = t >> 3, h = t & 7;
    const int si = self_idx[e], ni = nbr_idx[e];
    const float4* qv = (const float4*)(qp + (size_t)si * 128 + h * 16);
    const float4* kv = (const float4*)(kp + (size_t)ni * 128 + h * 16);
    float s = 0.0f;
#pragma unroll
    for (int j = 0; j < 4; ++j) {
        float4 a = qv[j], b = kv[j];
        s += a.x*b.x + a.y*b.y + a.z*b.z + a.w*b.w;
    }
    const float v = expf(s * 0.25f);      // 1/sqrt(HD=16)
    ex[t] = v;
    atomicAdd(&denom[(size_t)si * 8 + h], v);
}

// --------------- sum_e attn[e,h] per node (faithful rounding) --------------
__global__ void edge_attnsum_kernel(const float* __restrict__ ex,
                                    const float* __restrict__ denom,
                                    const int* __restrict__ self_idx,
                                    float* __restrict__ attnsum, int E)
{
    int t = blockIdx.x * blockDim.x + threadIdx.x;
    if (t >= E * 8) return;
    const int e = t >> 3, h = t & 7;
    const int si = self_idx[e];
    atomicAdd(&attnsum[(size_t)si * 8 + h], ex[t] / denom[(size_t)si * 8 + h]);
}

// y0[n][d] = attnsum[n][d/16] * vp[n][d]   (vp gathered with self == vp[n])
__global__ void scale_kernel(const float* __restrict__ attnsum,
                             const float* __restrict__ vp,
                             float* __restrict__ y0, int total)
{
    int t = blockIdx.x * blockDim.x + threadIdx.x;
    if (t >= total) return;
    const int n = t >> 7, d = t & 127;
    y0[t] = attnsum[(size_t)n * 8 + (d >> 4)] * vp[t];
}

// ------------------- out = LayerNorm(x + y) * w + b ------------------------
__global__ __launch_bounds__(256) void add_ln_kernel(
    const float* __restrict__ x, const float* __restrict__ y,
    const float* __restrict__ w, const float* __restrict__ b,
    float* __restrict__ out, int n)
{
    const int wave = threadIdx.x >> 5, lane = threadIdx.x & 31;
    const int row = blockIdx.x * 8 + wave;
    if (row >= n) return;
    const float4 xv = *(const float4*)(x + (size_t)row * 128 + lane * 4);
    const float4 yv = *(const float4*)(y + (size_t)row * 128 + lane * 4);
    float t0 = xv.x + yv.x, t1 = xv.y + yv.y, t2 = xv.z + yv.z, t3 = xv.w + yv.w;

    float s = t0 + t1 + t2 + t3;
#pragma unroll
    for (int off = 16; off; off >>= 1) s += __shfl_xor(s, off, 32);
    const float mu = s * (1.0f / 128.0f);

    const float d0 = t0 - mu, d1 = t1 - mu, d2 = t2 - mu, d3 = t3 - mu;
    float q = d0*d0 + d1*d1 + d2*d2 + d3*d3;
#pragma unroll
    for (int off = 16; off; off >>= 1) q += __shfl_xor(q, off, 32);
    const float inv = rsqrtf(q * (1.0f / 128.0f) + 1e-5f);

    const float4 wv = *(const float4*)(w + lane * 4);
    const float4 bv = *(const float4*)(b + lane * 4);
    float4 o;
    o.x = d0 * inv * wv.x + bv.x;
    o.y = d1 * inv * wv.y + bv.y;
    o.z = d2 * inv * wv.z + bv.z;
    o.w = d3 * inv * wv.w + bv.w;
    *(float4*)(out + (size_t)row * 128 + lane * 4) = o;
}

// ---------------------------------------------------------------------------
extern "C" void kernel_launch(void* const* d_in, const int* in_sizes, int n_in,
                              void* d_out, int out_size, void* d_ws, size_t ws_size,
                              hipStream_t stream)
{
    (void)n_in; (void)out_size; (void)ws_size;
    const float* q  = (const float*)d_in[0];
    const float* k  = (const float*)d_in[1];
    const float* v  = (const float*)d_in[2];
    const int* self_idx = (const int*)d_in[3];
    const int* nbr_idx  = (const int*)d_in[4];
    const float* Wq = (const float*)d_in[5];  const float* bq = (const float*)d_in[6];
    const float* Wk = (const float*)d_in[7];  const float* bk = (const float*)d_in[8];
    const float* Wv = (const float*)d_in[9];  const float* bv = (const float*)d_in[10];
    const float* Wo = (const float*)d_in[11]; const float* bo = (const float*)d_in[12];
    const float* W1 = (const float*)d_in[13]; const float* b1 = (const float*)d_in[14];
    const float* W2 = (const float*)d_in[15]; const float* b2 = (const float*)d_in[16];
    const float* ln1w = (const float*)d_in[17]; const float* ln1b = (const float*)d_in[18];
    const float* ln2w = (const float*)d_in[19]; const float* ln2b = (const float*)d_in[20];

    const int N = in_sizes[0] / 128;
    const int E = in_sizes[3];

    // ------------- workspace layout (chunks packed back to back, unpadded) -------------
    char* ws = (char*)d_ws;
    unsigned short* Wbf = (unsigned short*)ws;                 // 6 x 128x128 bf16
    size_t off = (size_t)6 * 16384 * sizeof(unsigned short);
    float* B1 = (float*)(ws + off); off += (size_t)N * 128 * 4;  // qp -> y0 -> h
    float* B2 = (float*)(ws + off); off += (size_t)N * 128 * 4;  // kp -> y1 -> m
    float* B3 = (float*)(ws + off); off += (size_t)N * 128 * 4;  // vp -> v1
    float* ex = (float*)(ws + off); off += (size_t)E * 8 * 4;
    float* denom   = (float*)(ws + off); off += (size_t)N * 8 * 4;
    float* attnsum = (float*)(ws + off); off += (size_t)N * 8 * 4;

    // ------------- weights to bf16 + zero accumulators each call ----------
    const float* Wsrc[6] = {Wq, Wk, Wv, Wo, W1, W2};
    for (int i = 0; i < 6; ++i)
        convert_bf16_kernel<<<64, 256, 0, stream>>>(Wsrc[i], Wbf + i * 16384, 16384);
    zero_kernel<<<(N * 16 + 255) / 256, 256, 0, stream>>>(denom, N * 16); // denom+attnsum

    const int gemm_grid = (N + 63) / 64;

    // ------------- Q/K/V projections (bf16 WMMA GEMMs) --------------------
    gemm_bias_kernel<<<gemm_grid, 256, 0, stream>>>(q, Wbf + 0 * 16384, bq, B1, N, 0);
    gemm_bias_kernel<<<gemm_grid, 256, 0, stream>>>(k, Wbf + 1 * 16384, bk, B2, N, 0);
    gemm_bias_kernel<<<gemm_grid, 256, 0, stream>>>(v, Wbf + 2 * 16384, bv, B3, N, 0);

    // ------------- edge attention (L2-resident gathers) -------------------
    const int eth = E * 8;
    edge_scores_kernel <<<(eth + 255) / 256, 256, 0, stream>>>(B1, B2, self_idx, nbr_idx, ex, denom, E);
    edge_attnsum_kernel<<<(eth + 255) / 256, 256, 0, stream>>>(ex, denom, self_idx, attnsum, E);

    // out_attn[n] = attnsum[n,h] * vp[n]  (reference gathers v with self_indices)
    scale_kernel<<<((N * 128) + 255) / 256, 256, 0, stream>>>(attnsum, B3, B1, N * 128);

    // ------------- output proj + residual/LN + MLP + residual/LN ----------
    gemm_bias_kernel<<<gemm_grid, 256, 0, stream>>>(B1, Wbf + 3 * 16384, bo, B2, N, 0);
    add_ln_kernel<<<(N + 7) / 8, 256, 0, stream>>>(v, B2, ln1w, ln1b, B3, N);
    gemm_bias_kernel<<<gemm_grid, 256, 0, stream>>>(B3, Wbf + 4 * 16384, b1, B1, N, 1);
    gemm_bias_kernel<<<gemm_grid, 256, 0, stream>>>(B1, Wbf + 5 * 16384, b2, B2, N, 0);
    add_ln_kernel<<<(N + 7) / 8, 256, 0, stream>>>(B3, B2, ln2w, ln2b, (float*)d_out, N);
}
